// SEMModel_7816840479014
// MI455X (gfx1250) — matrix operand order repaired and checked by reading, not compiled, not run on hardware
//
#include <hip/hip_runtime.h>
#include <math.h>

// ---------------------------------------------------------------------------
// CDNA5 (gfx1250) implementation of the complex Mamba-3 reference.
//   * All GEMMs: v_wmma_f32_16x16x32_bf16, operands pre-converted to bf16 in
//     global memory (weights pre-transposed to [N][K]) -> tiles stage to LDS
//     as raw b128 chunks, fragments load as ds_load_b128 (no packing moves).
//   * Tile staging uses GLOBAL_LOAD_ASYNC_TO_LDS_B128 + s_wait_asynccnt with
//     double buffering when the builtin exists; falls back to uint4 copies +
//     global_prefetch_b8 otherwise.
//   * (I + 0.05i*Hh) solve via 12-term Neumann series (spectral radius ~0.14).
//   * SSM scan: one wave32 per (b,d) channel, 2 complex states per lane.
// Workspace: ~105 MB fp32 + ~199 MB bf16 (~304 MB total).
// ---------------------------------------------------------------------------

typedef __attribute__((ext_vector_type(16))) __bf16 v16bf;
typedef __attribute__((ext_vector_type(8)))  float  v8f;
typedef int v4i __attribute__((vector_size(16)));

constexpr int   kD    = 1024;
constexpr int   kN    = 64;
constexpr int   kL    = 4;
constexpr int   kKL   = 4;
constexpr int   kV    = 32000;
constexpr int   kB    = 2;
constexpr int   kS    = 1024;
constexpr int   kBS   = kB * kS;
constexpr float kGAMMA = 0.01f;
constexpr float kDT    = 0.1f;
constexpr float kALPHA = 0.1f;
constexpr float kCOND  = 100.0f;
constexpr float kSCALE = 0.5f;   // 1/sqrt(L), L=4
constexpr float kEPS   = 1e-8f;
constexpr int   kNEUMANN = 12;   // 0.14^12 ~ 6e-11 << fp32 eps

// GEMM tiling: block = 256 thr = 8 waves (2Mx4N), wave = 16x32 (2 subtiles)
// block tile: M=32, N=128, K-step 32.
constexpr int kTK   = 32;
constexpr int kAOFF = 32 * 40;            // one A plane in LDS (ushorts)
constexpr int kBOFF = 128 * 40;           // one B plane in LDS
constexpr int kABUF = 2 * kAOFF;          // both A planes
constexpr int kBUF  = kABUF + 2 * kBOFF;  // full tile buffer (12800 ushorts)

#if defined(__has_builtin)
#if __has_builtin(__builtin_amdgcn_global_load_async_to_lds_b128)
#define HAVE_ASYNC 1
#endif
#endif

static __device__ __forceinline__ unsigned short f2bf(float f) {
  unsigned int u = __builtin_bit_cast(unsigned int, f);
  u += 0x7FFFu + ((u >> 16) & 1u);          // round-to-nearest-even
  return (unsigned short)(u >> 16);
}

#ifdef HAVE_ASYNC
static __device__ __forceinline__ void async_b128(const unsigned short* g,
                                                  unsigned short* l) {
  __builtin_amdgcn_global_load_async_to_lds_b128(
      (__attribute__((address_space(1))) v4i*)g,
      (__attribute__((address_space(3))) v4i*)l, 0, 0);
}
template <int CNT>
static __device__ __forceinline__ void wait_async() {
#if __has_builtin(__builtin_amdgcn_s_wait_asynccnt)
  __builtin_amdgcn_s_wait_asynccnt(CNT);
#else
  asm volatile("s_wait_asynccnt %0" :: "i"(CNT) : "memory");
#endif
}
#endif

// Stage one 32(M)x32(K) A tile + 128(N)x32(K) B tile (both planes, bf16) into
// LDS. 16B chunks: A = 256 chunks (1/thread), B = 1024 chunks (4/thread).
static __device__ __forceinline__ void stage_tile(
    const unsigned short* __restrict__ Ar, const unsigned short* __restrict__ Ai,
    const unsigned short* __restrict__ Br, const unsigned short* __restrict__ Bi,
    unsigned short* __restrict__ sb,
    int m0, int n0, int k0, int lda, int ldb, int tid)
{
  {
    int plane = tid >> 7, row = (tid >> 2) & 31, c = tid & 3;
    const unsigned short* src = (plane ? Ai : Ar) + (size_t)(m0 + row) * lda + k0 + c * 8;
    unsigned short* dst = sb + plane * kAOFF + row * 40 + c * 8;
#ifdef HAVE_ASYNC
    async_b128(src, dst);
#else
    *(uint4*)dst = *(const uint4*)src;
    __builtin_prefetch(src + kTK, 0, 0);
#endif
  }
#pragma unroll
  for (int i = 0; i < 4; ++i) {
    int ch = tid + i * 256;
    int plane = ch >> 9, row = (ch >> 2) & 127, c = ch & 3;
    const unsigned short* src = (plane ? Bi : Br) + (size_t)(n0 + row) * ldb + k0 + c * 8;
    unsigned short* dst = sb + kABUF + plane * kBOFF + row * 40 + c * 8;
#ifdef HAVE_ASYNC
    async_b128(src, dst);
#else
    *(uint4*)dst = *(const uint4*)src;
    __builtin_prefetch(src + kTK, 0, 0);
#endif
  }
}

// 8 WMMAs on the staged tile (A frag reused across 2 N-subtiles).
static __device__ __forceinline__ void compute_tile(
    const unsigned short* __restrict__ sb, int wm, int wn, int lane,
    v8f crr[2], v8f cii[2], v8f cri[2], v8f cir[2])
{
  union frag { v16bf v; uint4 q[2]; };
  const unsigned short* sAr = sb;
  const unsigned short* sAi = sb + kAOFF;
  const unsigned short* sBr = sb + kABUF;
  const unsigned short* sBi = sb + kABUF + kBOFF;

  const int arow = wm + (lane & 15);
  const int ahk  = (lane >> 4) << 4;     // lanes 16-31: K += 16
  frag fa_r, fa_i;
  fa_r.q[0] = *(const uint4*)(sAr + arow * 40 + ahk);
  fa_r.q[1] = *(const uint4*)(sAr + arow * 40 + 8 + ahk);
  fa_i.q[0] = *(const uint4*)(sAi + arow * 40 + ahk);
  fa_i.q[1] = *(const uint4*)(sAi + arow * 40 + 8 + ahk);

  const int bhk = (lane >> 4) << 4;      // lanes 16-31: K += 16
#pragma unroll
  for (int sub = 0; sub < 2; ++sub) {
    const int bcol = wn + sub * 16 + (lane & 15);
    frag fb_r, fb_i;
    fb_r.q[0] = *(const uint4*)(sBr + bcol * 40 + bhk);
    fb_r.q[1] = *(const uint4*)(sBr + bcol * 40 + bhk + 8);
    fb_i.q[0] = *(const uint4*)(sBi + bcol * 40 + bhk);
    fb_i.q[1] = *(const uint4*)(sBi + bcol * 40 + bhk + 8);
    crr[sub] = __builtin_amdgcn_wmma_f32_16x16x32_bf16(false, fa_r.v, false, fb_r.v, (short)0, crr[sub], false, false);
    cii[sub] = __builtin_amdgcn_wmma_f32_16x16x32_bf16(false, fa_i.v, false, fb_i.v, (short)0, cii[sub], false, false);
    cri[sub] = __builtin_amdgcn_wmma_f32_16x16x32_bf16(false, fa_r.v, false, fb_i.v, (short)0, cri[sub], false, false);
    cir[sub] = __builtin_amdgcn_wmma_f32_16x16x32_bf16(false, fa_i.v, false, fb_r.v, (short)0, cir[sub], false, false);
  }
}

// ---------------------------------------------------------------------------
// Complex GEMM, A[M,K] bf16 row-major, B[N,K] bf16 row-major (i.e. B^T mult).
//   CONJB: conjugate B;   EPI: 0 store (Cr,Ci), 1 store |C|^2, 2 store X + C
// ---------------------------------------------------------------------------
enum { EPI_STORE = 0, EPI_ABS2 = 1, EPI_ADDX = 2 };

template <bool CONJB, int EPI>
__global__ __launch_bounds__(256) void cgemm_nt(
    const unsigned short* __restrict__ Ar, const unsigned short* __restrict__ Ai,
    const unsigned short* __restrict__ Br, const unsigned short* __restrict__ Bi,
    float* __restrict__ Cr, float* __restrict__ Ci,
    const float* __restrict__ Xr, const float* __restrict__ Xi,
    int M, int N, int K, int lda, int ldb, int ldc)
{
#ifdef HAVE_ASYNC
  __shared__ unsigned short smem[2 * kBUF];
#else
  __shared__ unsigned short smem[kBUF];
#endif
  const int tid  = threadIdx.x;
  const int lane = tid & 31;
  const int wave = tid >> 5;
  const int m0   = blockIdx.y * 32;
  const int n0   = blockIdx.x * 128;
  const int wm   = (wave >> 2) << 4;   // 0 / 16
  const int wn   = (wave & 3) << 5;    // 0 / 32 / 64 / 96

  v8f z8 = {0,0,0,0,0,0,0,0};
  v8f crr[2] = {z8, z8}, cii[2] = {z8, z8}, cri[2] = {z8, z8}, cir[2] = {z8, z8};

  const int KT = K / kTK;
#ifdef HAVE_ASYNC
  stage_tile(Ar, Ai, Br, Bi, smem, m0, n0, 0, lda, ldb, tid);
  for (int kt = 0; kt < KT; ++kt) {
    unsigned short* cur = smem + (kt & 1) * kBUF;
    if (kt + 1 < KT) {
      stage_tile(Ar, Ai, Br, Bi, smem + ((kt + 1) & 1) * kBUF,
                 m0, n0, (kt + 1) * kTK, lda, ldb, tid);
      wait_async<5>();                 // current tile's 5 DMAs done; next in flight
    } else {
      wait_async<0>();
    }
    __syncthreads();
    compute_tile(cur, wm, wn, lane, crr, cii, cri, cir);
    __syncthreads();                   // all reads done before buffer reuse
  }
#else
  for (int kt = 0; kt < KT; ++kt) {
    __syncthreads();
    stage_tile(Ar, Ai, Br, Bi, smem, m0, n0, kt * kTK, lda, ldb, tid);
    __syncthreads();
    compute_tile(smem, wm, wn, lane, crr, cii, cri, cir);
  }
#endif

  const int crow = wm + ((lane >> 4) << 3);
#pragma unroll
  for (int sub = 0; sub < 2; ++sub) {
    const int ccol = n0 + wn + sub * 16 + (lane & 15);
#pragma unroll
    for (int r = 0; r < 8; ++r) {
      int m = m0 + crow + r;
      size_t o = (size_t)m * ldc + ccol;
      float vr = CONJB ? crr[sub][r] + cii[sub][r] : crr[sub][r] - cii[sub][r];
      float vi = CONJB ? cir[sub][r] - cri[sub][r] : cri[sub][r] + cir[sub][r];
      if (EPI == EPI_STORE)      { Cr[o] = vr; Ci[o] = vi; }
      else if (EPI == EPI_ABS2)  { Cr[o] = vr * vr + vi * vi; }
      else                       { Cr[o] = Xr[o] + vr; Ci[o] = Xi[o] + vi; }
    }
  }
}

// ---------------------------------------------------------------------------
// bf16 conversion / transpose kernels
// ---------------------------------------------------------------------------
__global__ __launch_bounds__(256) void k_f2bf(
    const float* __restrict__ in, unsigned short* __restrict__ out, long n)
{
  long idx = (long)blockIdx.x * 256 + threadIdx.x;
  if (idx < n) out[idx] = f2bf(in[idx]);
}

// out[n*K+k] = bf16(in[k*N+n])  (LDS-tiled transpose)
__global__ __launch_bounds__(256) void k_trbf(
    const float* __restrict__ in, unsigned short* __restrict__ out, int K, int N)
{
  __shared__ float tile[32][33];
  int k0 = blockIdx.y * 32, n0 = blockIdx.x * 32;
  int tx = threadIdx.x & 31, ty = threadIdx.x >> 5;   // 32 x 8
#pragma unroll
  for (int i = 0; i < 32; i += 8)
    tile[ty + i][tx] = in[(size_t)(k0 + ty + i) * N + n0 + tx];
  __syncthreads();
#pragma unroll
  for (int i = 0; i < 32; i += 8)
    out[(size_t)(n0 + ty + i) * K + k0 + tx] = f2bf(tile[tx][ty + i]);
}

// ---------------------------------------------------------------------------
// Elementwise / small kernels
// ---------------------------------------------------------------------------
__global__ __launch_bounds__(256) void k_embed(
    const float* __restrict__ er, const float* __restrict__ ei,
    const int* __restrict__ ids, float* __restrict__ xr, float* __restrict__ xi)
{
  int idx = blockIdx.x * 256 + threadIdx.x;       // over kBS*kD
  int bs = idx >> 10, d = idx & (kD - 1);
  int tok = ids[bs];
  xr[idx] = er[(size_t)tok * kD + d];
  xi[idx] = ei[(size_t)tok * kD + d];
}

__global__ __launch_bounds__(256) void k_conv(
    const float* __restrict__ uzr, const float* __restrict__ uzi,
    const float* __restrict__ wr,  const float* __restrict__ wi,
    float* __restrict__ ur, float* __restrict__ ui)
{
  int idx = blockIdx.x * 256 + threadIdx.x;       // (b*S+s)*D + d
  int d = idx & (kD - 1);
  int s = (idx >> 10) & (kS - 1);
  int b = idx >> 20;
  float ar = 0.f, ai = 0.f;
#pragma unroll
  for (int k = 0; k < 4; ++k) {
    int sp = s - k;
    if (sp >= 0) {
      size_t g = ((size_t)(b * kS + sp)) * (2 * kD) + d;
      float xr_ = uzr[g], xi_ = uzi[g];
      float cr = wr[d * 4 + k], ci = wi[d * 4 + k];
      ar += cr * xr_ - ci * xi_;
      ai += cr * xi_ + ci * xr_;
    }
  }
  ur[idx] = ar; ui[idx] = ai;
}

// SSM scan: one wave per (b,d) channel; lane holds states n=lane, n=lane+32.
__global__ __launch_bounds__(256) void k_scan(
    const float* __restrict__ ur, const float* __restrict__ ui,
    const float* __restrict__ uzr, const float* __restrict__ uzi,
    const float* __restrict__ alr, const float* __restrict__ aim,
    const float* __restrict__ bmr, const float* __restrict__ bmi,
    const float* __restrict__ cmr, const float* __restrict__ cmi,
    float* __restrict__ yr, float* __restrict__ yi)
{
  int ch   = blockIdx.x * 8 + (threadIdx.x >> 5);
  int lane = threadIdx.x & 31;
  int b = ch >> 10, d = ch & (kD - 1);
  int base = d * kN;
  int n0 = base + lane, n1 = base + lane + 32;

  float m0 = expf(-expf(alr[n0])), a0 = aim[n0];
  float m1 = expf(-expf(alr[n1])), a1 = aim[n1];
  float d0r = m0 * cosf(a0), d0i = m0 * sinf(a0);
  float d1r = m1 * cosf(a1), d1i = m1 * sinf(a1);
  float b0r = bmr[n0], b0i = bmi[n0], b1r = bmr[n1], b1i = bmi[n1];
  float c0r = cmr[n0], c0i = cmi[n0], c1r = cmr[n1], c1i = cmi[n1];

  float h0r = 0.f, h0i = 0.f, h1r = 0.f, h1i = 0.f;
  for (int s = 0; s < kS; ++s) {
    size_t rowu = ((size_t)(b * kS + s)) * kD + d;
    float utr = ur[rowu], uti = ui[rowu];
    float t0r = d0r * h0r - d0i * h0i + b0r * utr - b0i * uti;
    float t0i = d0r * h0i + d0i * h0r + b0r * uti + b0i * utr;
    float t1r = d1r * h1r - d1i * h1i + b1r * utr - b1i * uti;
    float t1i = d1r * h1i + d1i * h1r + b1r * uti + b1i * utr;
    h0r = t0r; h0i = t0i; h1r = t1r; h1i = t1i;
    float pr = c0r * h0r - c0i * h0i + c1r * h1r - c1i * h1i;
    float pi = c0r * h0i + c0i * h0r + c1r * h1i + c1i * h1r;
#pragma unroll
    for (int off = 16; off > 0; off >>= 1) {
      pr += __shfl_xor(pr, off, 32);
      pi += __shfl_xor(pi, off, 32);
    }
    if (lane == 0) {
      size_t rowz = ((size_t)(b * kS + s)) * (2 * kD) + kD + d;
      float zr = uzr[rowz], zi = uzi[rowz];
      float sg = 1.f / (1.f + expf(-zr));
      float zsr = zr * sg, zsi = zi * sg;
      yr[rowu] = pr * zsr - pi * zsi;
      yi[rowu] = pr * zsi + pi * zsr;
    }
  }
}

__global__ __launch_bounds__(256) void k_lproj(
    const float* __restrict__ lr, const float* __restrict__ li,
    const float* __restrict__ br_, const float* __restrict__ bi_,
    float* __restrict__ pr_, float* __restrict__ pi_)
{
  int idx = blockIdx.x * 256 + threadIdx.x;   // kBS*kKL
  int bs = idx >> 2, k = idx & 3;
  const float* rr = br_ + (size_t)bs * kD;
  const float* ri = bi_ + (size_t)bs * kD;
  float ar = 0.f, ai = 0.f;
  for (int d = 0; d < kD; ++d) {
    float wr = lr[k * kD + d], wi = li[k * kD + d];
    float xr = rr[d], xi = ri[d];
    ar += wr * xr + wi * xi;          // conj(w) * x
    ai += wr * xi - wi * xr;
  }
  pr_[idx] = ar; pi_[idx] = ai;
}

__global__ __launch_bounds__(256) void k_lupd(
    const float* __restrict__ lr, const float* __restrict__ li,
    const float* __restrict__ pr_, const float* __restrict__ pi_,
    float* __restrict__ br_, float* __restrict__ bi_)
{
  int idx = blockIdx.x * 256 + threadIdx.x;   // kBS*kD
  int bs = idx >> 10, d = idx & (kD - 1);
  float ar = 0.f, ai = 0.f;
#pragma unroll
  for (int k = 0; k < kKL; ++k) {
    float wr = lr[k * kD + d], wi = li[k * kD + d];
    float qr = pr_[bs * kKL + k], qi = pi_[bs * kKL + k];
    ar += wr * qr - wi * qi;
    ai += wr * qi + wi * qr;
  }
  br_[idx] -= kGAMMA * ar;
  bi_[idx] -= kGAMMA * ai;
}

__global__ __launch_bounds__(256) void k_diag(
    const float* __restrict__ br_, const float* __restrict__ bi_,
    float* __restrict__ dg)
{
  int idx = blockIdx.x * 256 + threadIdx.x;   // kB*kD
  int b = idx >> 10, d = idx & (kD - 1);
  float s = 0.f;
  for (int t = 0; t < kS; ++t) {
    size_t g = ((size_t)(b * kS + t)) * kD + d;
    float a = br_[g], c = bi_[g];
    s += a * a + c * c;
  }
  dg[idx] = s / (float)kS;
}

__global__ __launch_bounds__(256) void k_cond(
    const float* __restrict__ dg, const float* __restrict__ theta, int l,
    float* __restrict__ pf)
{
  int b = blockIdx.x, t = threadIdx.x;
  __shared__ float smx[256], smn[256];
  float mx = -1e30f, mn = 1e30f;
  for (int d = t; d < kD; d += 256) {
    float v = dg[b * kD + d];
    mx = fmaxf(mx, v); mn = fminf(mn, v);
  }
  smx[t] = mx; smn[t] = mn; __syncthreads();
  for (int o = 128; o > 0; o >>= 1) {
    if (t < o) { smx[t] = fmaxf(smx[t], smx[t + o]); smn[t] = fminf(smn[t], smn[t + o]); }
    __syncthreads();
  }
  if (t == 0) {
    float cond = smx[0] / (smn[0] + kEPS);
    if (cond > kCOND) { float th = theta[l]; pf[2 * b] = cosf(th); pf[2 * b + 1] = sinf(th); }
    else              { pf[2 * b] = 1.f; pf[2 * b + 1] = 0.f; }
  }
}

__global__ __launch_bounds__(256) void k_resid(
    const float* __restrict__ br_, const float* __restrict__ bi_,
    const float* __restrict__ pf, float* __restrict__ xr, float* __restrict__ xi)
{
  int idx = blockIdx.x * 256 + threadIdx.x;   // kBS*kD
  int b = idx >> 20;
  float pr = pf[2 * b], pi = pf[2 * b + 1];
  float ar = br_[idx], ai = bi_[idx];
  xr[idx] += kSCALE * (ar * pr - ai * pi);
  xi[idx] += kSCALE * (ar * pi + ai * pr);
}

// Aneg = -(0.5j*DT*Hh), Hh = 0.5*(P + P^H)
__global__ __launch_bounds__(256) void k_buildA(
    const float* __restrict__ Pr, const float* __restrict__ Pi,
    float* __restrict__ Anr, float* __restrict__ Ani)
{
  int idx = blockIdx.x * 256 + threadIdx.x;   // kD*kD
  int d = idx >> 10, e = idx & (kD - 1);
  float hr = 0.5f * (Pr[(size_t)d * kD + e] + Pr[(size_t)e * kD + d]);
  float hi = 0.5f * (Pi[(size_t)d * kD + e] - Pi[(size_t)e * kD + d]);
  const float c = 0.5f * kDT;
  Anr[idx] =  c * hi;   // real(-i*c*Hh)
  Ani[idx] = -c * hr;   // imag(-i*c*Hh)
}

__global__ __launch_bounds__(256) void k_init2(
    const float* __restrict__ wr, const float* __restrict__ wi,
    float* __restrict__ yr, float* __restrict__ yi,
    float* __restrict__ tr, float* __restrict__ ti)
{
  int idx = blockIdx.x * 256 + threadIdx.x;
  float a = wr[idx], b = wi[idx];
  yr[idx] = a; yi[idx] = b; tr[idx] = a; ti[idx] = b;
}

__global__ __launch_bounds__(256) void k_axpy(
    const float* __restrict__ tr, const float* __restrict__ ti,
    float* __restrict__ yr, float* __restrict__ yi)
{
  int idx = blockIdx.x * 256 + threadIdx.x;
  yr[idx] += tr[idx];
  yi[idx] += ti[idx];
}

// phase twist + RMS norm, one workgroup per (b,s) row
__global__ __launch_bounds__(256) void k_phasenorm(
    const float* __restrict__ yr, const float* __restrict__ yi,
    const float* __restrict__ nw, float* __restrict__ outr, float* __restrict__ outi)
{
  int row = blockIdx.x, t = threadIdx.x;
  __shared__ float red[256];
  const float* rr = yr + (size_t)row * kD;
  const float* ri = yi + (size_t)row * kD;
  float s = 0.f;
  for (int d = t; d < kD; d += 256) { float a = rr[d], b = ri[d]; s += a * a + b * b; }
  red[t] = s; __syncthreads();
  for (int o = 128; o > 0; o >>= 1) {
    if (t < o) red[t] += red[t + o];
    __syncthreads();
  }
  float inv = 1.f / sqrtf(red[0] / (float)kD + kEPS);
  for (int d = t; d < kD; d += 256) {
    float a = rr[d], b = ri[d];
    float ph = kALPHA * (a * a + b * b);
    float cs = cosf(ph), sn = sinf(ph);
    float w = nw[d] * inv;
    outr[(size_t)row * kD + d] = (a * cs - b * sn) * w;
    outi[(size_t)row * kD + d] = (a * sn + b * cs) * w;
  }
}

// ---------------------------------------------------------------------------
extern "C" void kernel_launch(void* const* d_in, const int* in_sizes, int n_in,
                              void* d_out, int out_size, void* d_ws, size_t ws_size,
                              hipStream_t stream)
{
  (void)in_sizes; (void)n_in; (void)out_size; (void)ws_size;
  const float* emb_r     = (const float*)d_in[0];
  const float* emb_i     = (const float*)d_in[1];
  const float* in_proj_r = (const float*)d_in[2];
  const float* in_proj_i = (const float*)d_in[3];
  const float* conv_w_r  = (const float*)d_in[4];
  const float* conv_w_i  = (const float*)d_in[5];
  const float* A_logr    = (const float*)d_in[6];
  const float* A_imag    = (const float*)d_in[7];
  const float* Bm_r      = (const float*)d_in[8];
  const float* Bm_i      = (const float*)d_in[9];
  const float* Cm_r      = (const float*)d_in[10];
  const float* Cm_i      = (const float*)d_in[11];
  const float* out_proj_r= (const float*)d_in[12];
  const float* out_proj_i= (const float*)d_in[13];
  const float* lind_r    = (const float*)d_in[14];
  const float* lind_i    = (const float*)d_in[15];
  const float* quat_theta= (const float*)d_in[16];
  const float* prop_H_r  = (const float*)d_in[17];
  const float* prop_H_i  = (const float*)d_in[18];
  const float* norm_w    = (const float*)d_in[19];
  const float* W_out_r   = (const float*)d_in[20];
  const float* W_out_i   = (const float*)d_in[21];
  const int*   token_ids = (const int*)d_in[22];
  float* out = (float*)d_out;

  // ---- fp32 workspace ----
  float* ws = (float*)d_ws;
  size_t off = 0;
  auto F = [&](size_t n) { float* p = ws + off; off += n; return p; };
  const size_t P  = (size_t)kBS * kD;         // 2M
  const size_t DD = (size_t)kD * kD;          // 1M
  float* xr  = F(P);     float* xi  = F(P);
  float* uzr = F(2 * P); float* uzi = F(2 * P);
  float* ucr = F(P);     float* uci = F(P);
  float* ygr = F(P);     float* ygi = F(P);
  float* brr = F(P);     float* bri = F(P);
  float* Anr = F(DD);    float* Ani = F(DD);
  float* prr = F((size_t)kBS * kKL); float* pri = F((size_t)kBS * kKL);
  float* dg  = F((size_t)kB * kD);
  float* pf  = F(2 * kB);

  // ---- bf16 workspace (16B-aligned; all sizes multiples of 8) ----
  off = (off + 7) & ~(size_t)7;
  unsigned short* us = (unsigned short*)(ws + off);
  size_t uo = 0;
  auto U = [&](size_t n) { unsigned short* p = us + uo; uo += n; return p; };
  unsigned short* inwTr = U(kL * 2 * DD); unsigned short* inwTi = U(kL * 2 * DD);
  unsigned short* owTr  = U(kL * DD);     unsigned short* owTi  = U(kL * DD);
  unsigned short* wvr   = U((size_t)kV * kD); unsigned short* wvi = U((size_t)kV * kD);
  unsigned short* abnr  = U(DD);          unsigned short* abni  = U(DD);
  unsigned short* xbr   = U(P);           unsigned short* xbi   = U(P);
  unsigned short* abr   = U(P);           unsigned short* abi   = U(P);

  const dim3 blk(256);
  const unsigned gP = (unsigned)(P / 256);

  // ---- weight prep: transpose+convert to bf16 [N][K] ----
  for (int l = 0; l < kL; ++l) {
    k_trbf<<<dim3(2 * kD / 32, kD / 32), blk, 0, stream>>>(
        in_proj_r + (size_t)l * 2 * DD, inwTr + (size_t)l * 2 * DD, kD, 2 * kD);
    k_trbf<<<dim3(2 * kD / 32, kD / 32), blk, 0, stream>>>(
        in_proj_i + (size_t)l * 2 * DD, inwTi + (size_t)l * 2 * DD, kD, 2 * kD);
    k_trbf<<<dim3(kD / 32, kD / 32), blk, 0, stream>>>(
        out_proj_r + (size_t)l * DD, owTr + (size_t)l * DD, kD, kD);
    k_trbf<<<dim3(kD / 32, kD / 32), blk, 0, stream>>>(
        out_proj_i + (size_t)l * DD, owTi + (size_t)l * DD, kD, kD);
  }
  k_f2bf<<<(unsigned)((size_t)kV * kD / 256), blk, 0, stream>>>(W_out_r, wvr, (long)kV * kD);
  k_f2bf<<<(unsigned)((size_t)kV * kD / 256), blk, 0, stream>>>(W_out_i, wvi, (long)kV * kD);

  // ---- embedding ----
  k_embed<<<gP, blk, 0, stream>>>(emb_r, emb_i, token_ids, xr, xi);

  // ---- layers ----
  for (int l = 0; l < kL; ++l) {
    const float* cwr = conv_w_r + (size_t)l * kD * 4;
    const float* cwi = conv_w_i + (size_t)l * kD * 4;
    const float* alr = A_logr   + (size_t)l * kD * kN;
    const float* aim = A_imag   + (size_t)l * kD * kN;
    const float* bmr = Bm_r + (size_t)l * kD * kN; const float* bmi = Bm_i + (size_t)l * kD * kN;
    const float* cmr = Cm_r + (size_t)l * kD * kN; const float* cmi = Cm_i + (size_t)l * kD * kN;
    const float* ldr = lind_r + (size_t)l * kKL * kD;
    const float* ldi = lind_i + (size_t)l * kKL * kD;

    k_f2bf<<<gP, blk, 0, stream>>>(xr, xbr, (long)P);
    k_f2bf<<<gP, blk, 0, stream>>>(xi, xbi, (long)P);
    cgemm_nt<false, EPI_STORE><<<dim3(2 * kD / 128, kBS / 32), blk, 0, stream>>>(
        xbr, xbi, inwTr + (size_t)l * 2 * DD, inwTi + (size_t)l * 2 * DD,
        uzr, uzi, nullptr, nullptr, kBS, 2 * kD, kD, kD, kD, 2 * kD);
    k_conv<<<gP, blk, 0, stream>>>(uzr, uzi, cwr, cwi, ucr, uci);
    k_scan<<<kB * kD / 8, blk, 0, stream>>>(ucr, uci, uzr, uzi, alr, aim,
                                            bmr, bmi, cmr, cmi, ygr, ygi);
    k_f2bf<<<gP, blk, 0, stream>>>(ygr, abr, (long)P);
    k_f2bf<<<gP, blk, 0, stream>>>(ygi, abi, (long)P);
    cgemm_nt<false, EPI_STORE><<<dim3(kD / 128, kBS / 32), blk, 0, stream>>>(
        abr, abi, owTr + (size_t)l * DD, owTi + (size_t)l * DD,
        brr, bri, nullptr, nullptr, kBS, kD, kD, kD, kD, kD);
    k_lproj<<<kBS * kKL / 256, blk, 0, stream>>>(ldr, ldi, brr, bri, prr, pri);
    k_lupd<<<gP, blk, 0, stream>>>(ldr, ldi, prr, pri, brr, bri);
    k_diag<<<kB * kD / 256, blk, 0, stream>>>(brr, bri, dg);
    k_cond<<<kB, blk, 0, stream>>>(dg, quat_theta, l, pf);
    k_resid<<<gP, blk, 0, stream>>>(brr, bri, pf, xr, xi);
  }

  // ---- Cayley propagator via Neumann series ----
  k_buildA<<<(unsigned)(DD / 256), blk, 0, stream>>>(prop_H_r, prop_H_i, Anr, Ani);
  k_f2bf<<<(unsigned)(DD / 256), blk, 0, stream>>>(Anr, abnr, (long)DD);
  k_f2bf<<<(unsigned)(DD / 256), blk, 0, stream>>>(Ani, abni, (long)DD);
  k_f2bf<<<gP, blk, 0, stream>>>(xr, xbr, (long)P);
  k_f2bf<<<gP, blk, 0, stream>>>(xi, xbi, (long)P);

  float* Wr  = uzr;  float* Wi  = uzr + P;   // reuse layer scratch
  float* Tar = uzi;  float* Tai = uzi + P;
  float* Tbr = ucr;  float* Tbi = uci;
  float* Yr  = ygr;  float* Yi  = ygi;
  cgemm_nt<false, EPI_ADDX><<<dim3(kD / 128, kBS / 32), blk, 0, stream>>>(
      xbr, xbi, abnr, abni, Wr, Wi, xr, xi, kBS, kD, kD, kD, kD, kD);
  k_init2<<<gP, blk, 0, stream>>>(Wr, Wi, Yr, Yi, Tar, Tai);
  float* tir = Tar; float* tii = Tai; float* tor_ = Tbr; float* toi = Tbi;
  for (int it = 0; it < kNEUMANN; ++it) {
    k_f2bf<<<gP, blk, 0, stream>>>(tir, abr, (long)P);
    k_f2bf<<<gP, blk, 0, stream>>>(tii, abi, (long)P);
    cgemm_nt<false, EPI_STORE><<<dim3(kD / 128, kBS / 32), blk, 0, stream>>>(
        abr, abi, abnr, abni, tor_, toi, nullptr, nullptr, kBS, kD, kD, kD, kD, kD);
    k_axpy<<<gP, blk, 0, stream>>>(tor_, toi, Yr, Yi);
    float* t;
    t = tir; tir = tor_; tor_ = t;
    t = tii; tii = toi;  toi  = t;
  }

  // ---- nonlinear phase + RMS norm ----
  k_phasenorm<<<kBS, blk, 0, stream>>>(Yr, Yi, norm_w, brr, bri);
  k_f2bf<<<gP, blk, 0, stream>>>(brr, abr, (long)P);
  k_f2bf<<<gP, blk, 0, stream>>>(bri, abi, (long)P);

  // ---- vocab projection: |y . conj(W_out)^T|^2 ----
  cgemm_nt<true, EPI_ABS2><<<dim3(kV / 128, kBS / 32), blk, 0, stream>>>(
      abr, abi, wvr, wvi, out, nullptr, nullptr, nullptr,
      kBS, kV, kD, kD, kD, kV);
}
